// RaggedTopKGatingModule_62216896250151
// MI455X (gfx1250) — compile-verified, matrix operand order checked by reading, not run on hardware
//
#include <hip/hip_runtime.h>
#include <hip/hip_bf16.h>
#include <math.h>

#define N_TOKENS   524288
#define N_EXPERTS  64
#define TOP_K      8
#define NSLOTS     (N_TOKENS * TOP_K)          // 4194304
#define CHUNK_TOK  128
#define CHUNK_SLOT (CHUNK_TOK * TOP_K)         // 1024
#define NCHUNKS    (N_TOKENS / CHUNK_TOK)      // 4096

// d_out element offsets (all float32): counts | scores | assign | offs | logits
#define OFF_COUNTS 0
#define OFF_SCORES 64
#define OFF_ASSIGN (64 + NSLOTS)
#define OFF_OFFS   (64 + 2 * NSLOTS)
#define OFF_LOGITS (64 + 3 * NSLOTS)

typedef __attribute__((ext_vector_type(8))) int v8i;

// ---------------------------------------------------------------------------
// K1: per-token softmax + top-8.  One wave32 per token (2 logits per lane).
// ---------------------------------------------------------------------------
__global__ __launch_bounds__(256)
void k1_gate(const float* __restrict__ logits, float* __restrict__ out) {
    const int lane  = threadIdx.x & 31;
    const int wave  = threadIdx.x >> 5;
    const int token = blockIdx.x * 8 + wave;

    const float* lp = logits + (size_t)token * 64;
    const float l0 = lp[lane];
    const float l1 = lp[lane + 32];

    float w0 = l0, w1 = l1;
    float maxv = 0.0f;
    float myv = 0.0f;
    int   myi = 0;

#pragma unroll
    for (int k = 0; k < TOP_K; ++k) {
        float bv = w0; int bi = lane;
        if (w1 > bv) { bv = w1; bi = lane + 32; }   // tie -> keep lower index
#pragma unroll
        for (int off = 16; off > 0; off >>= 1) {
            float ov = __shfl_xor(bv, off, 32);
            int   oi = __shfl_xor(bi, off, 32);
            if (ov > bv || (ov == bv && oi < bi)) { bv = ov; bi = oi; }
        }
        if (k == 0) maxv = bv;
        if (lane == k) { myv = bv; myi = bi; }
        if (bi == lane)           w0 = -INFINITY;
        else if (bi == lane + 32) w1 = -INFINITY;
    }

    // softmax denominator over all 64 (uses unmasked copies)
    float s = expf(l0 - maxv) + expf(l1 - maxv);
#pragma unroll
    for (int off = 16; off > 0; off >>= 1) s += __shfl_xor(s, off, 32);
    const float inv = 1.0f / s;

    if (lane < TOP_K) {
        const size_t p = (size_t)token * TOP_K + lane;
        out[OFF_SCORES + p] = expf(myv - maxv) * inv;
        out[OFF_ASSIGN + p] = (float)myi;
    }
}

// ---------------------------------------------------------------------------
// K2: per-chunk expert histogram via WMMA one-hot matmul.
//     A (16x64 iu8) = one-hot(expert id == e_base + row), B = ones.
//     D[m][n] = count(e_base + m) over 64 slots; accumulate over slot groups.
// ---------------------------------------------------------------------------
__global__ __launch_bounds__(256)
void k2_hist(const float* __restrict__ outRO, int* __restrict__ partial) {
    __shared__ unsigned ids[CHUNK_SLOT / 4];   // 256 dwords = 1024 expert bytes
    __shared__ int hist[N_EXPERTS];

    const int t     = threadIdx.x;
    const int chunk = blockIdx.x;

    const float* af = outRO + OFF_ASSIGN + (size_t)chunk * CHUNK_SLOT;
    unsigned b0 = (unsigned)(int)af[t * 4 + 0];
    unsigned b1 = (unsigned)(int)af[t * 4 + 1];
    unsigned b2 = (unsigned)(int)af[t * 4 + 2];
    unsigned b3 = (unsigned)(int)af[t * 4 + 3];
    ids[t] = b0 | (b1 << 8) | (b2 << 16) | (b3 << 24);
    if (t < N_EXPERTS) hist[t] = 0;
    __syncthreads();

    const int lane = t & 31, wave = t >> 5;
    const int m = lane & 15, half = lane >> 4;

    v8i B;
#pragma unroll
    for (int j = 0; j < 8; ++j) B[j] = 0x01010101;   // all-ones (layout-invariant)

#pragma unroll
    for (int ebi = 0; ebi < 4; ++ebi) {
        const int eb = ebi * 16;
        const unsigned pat = (unsigned)(eb + m) * 0x01010101u;
        v8i C = {0, 0, 0, 0, 0, 0, 0, 0};
#pragma unroll
        for (int gi = 0; gi < 2; ++gi) {
            const int g = wave * 2 + gi;             // 64-slot group in chunk
            v8i A;
#pragma unroll
            for (int v = 0; v < 8; ++v) {
                // documented 8-bit A 16x64 layout
                const int kb = half * 8 + (v & 1) * 4 + ((v >> 1) & 3) * 16;
                const unsigned x = ids[g * 16 + (kb >> 2)] ^ pat;
                // exact zero-byte -> 0x01 per matching byte (no carry aliasing)
                const unsigned oh =
                    (~(((x & 0x7F7F7F7Fu) + 0x7F7F7F7Fu) | x | 0x7F7F7F7Fu)) >> 7;
                A[v] = (int)oh;
            }
            C = __builtin_amdgcn_wmma_i32_16x16x64_iu8(false, A, false, B, C,
                                                       false, false);
        }
        // D[m][n] identical across n; lane 0 holds rows 0..7 (VGPR j -> M=j),
        // lane 16 holds rows 8..15.
        if (lane == 0 || lane == 16) {
#pragma unroll
            for (int j = 0; j < 8; ++j)
                atomicAdd(&hist[eb + half * 8 + j], C[j]);
        }
    }
    __syncthreads();
    if (t < N_EXPERTS) partial[chunk * N_EXPERTS + t] = hist[t];
}

// ---------------------------------------------------------------------------
// K3: per-expert exclusive scan over 4096 chunk partials; totals -> counts.
//     One block per expert; 256 threads x 16 chunks each.
// ---------------------------------------------------------------------------
__global__ __launch_bounds__(256)
void k3_scan(const int* __restrict__ partial, int* __restrict__ base,
             float* __restrict__ out) {
    __shared__ int sh[256];
    const int e = blockIdx.x;
    const int t = threadIdx.x;

    int p[16];
    int tot = 0;
#pragma unroll
    for (int i = 0; i < 16; ++i) {
        p[i] = partial[(size_t)(t * 16 + i) * N_EXPERTS + e];
        tot += p[i];
    }
    sh[t] = tot;
    __syncthreads();
    for (int off = 1; off < 256; off <<= 1) {
        int v = (t >= off) ? sh[t - off] : 0;
        __syncthreads();
        sh[t] += v;
        __syncthreads();
    }
    int run = sh[t] - tot;                    // exclusive prefix
#pragma unroll
    for (int i = 0; i < 16; ++i) {
        base[(size_t)(t * 16 + i) * N_EXPERTS + e] = run;
        run += p[i];
    }
    if (t == 255) out[OFF_COUNTS + e] = (float)run;   // global count
}

// ---------------------------------------------------------------------------
// K4: intra-chunk deterministic ranks. One wave32 per chunk; sequential
//     32-slot groups; equal-expert lane mask via 6 AND-ed ballots.
// ---------------------------------------------------------------------------
__global__ __launch_bounds__(32)
void k4_offsets(const float* __restrict__ outRO, const int* __restrict__ base,
                float* __restrict__ out) {
    __shared__ int cnt[N_EXPERTS];
    const int lane  = threadIdx.x;
    const int chunk = blockIdx.x;

    cnt[lane]      = base[chunk * N_EXPERTS + lane];
    cnt[lane + 32] = base[chunk * N_EXPERTS + lane + 32];
    __syncthreads();

    const float* af = outRO + OFF_ASSIGN + (size_t)chunk * CHUNK_SLOT;
    float*       of = out   + OFF_OFFS   + (size_t)chunk * CHUNK_SLOT;
    const unsigned ltmask = (1u << lane) - 1u;

    for (int g = 0; g < 32; ++g) {
        const int e = (int)af[g * 32 + lane];
        unsigned msk = 0xffffffffu;
#pragma unroll
        for (int b = 0; b < 6; ++b) {
            const bool bit = (e >> b) & 1;
            const unsigned bal = (unsigned)__ballot(bit);
            msk &= bit ? bal : ~bal;
        }
        const int cnte   = __popc(msk);
        const int rank   = __popc(msk & ltmask);
        const int leader = __ffs((int)msk) - 1;
        int bv = 0;
        if (lane == leader) bv = atomicAdd(&cnt[e], cnte);  // LDS, in-order per wave
        bv = __shfl(bv, leader, 32);
        of[g * 32 + lane] = (float)(bv + rank);
    }
}

// ---------------------------------------------------------------------------
// K5: logits pass-through copy (float4, grid-stride).
// ---------------------------------------------------------------------------
__global__ __launch_bounds__(256)
void k5_copy(const float4* __restrict__ src, float4* __restrict__ dst, int n4) {
    int i = blockIdx.x * blockDim.x + threadIdx.x;
    const int stride = gridDim.x * blockDim.x;
    for (; i < n4; i += stride) dst[i] = src[i];
}

// ---------------------------------------------------------------------------
extern "C" void kernel_launch(void* const* d_in, const int* in_sizes, int n_in,
                              void* d_out, int out_size, void* d_ws, size_t ws_size,
                              hipStream_t stream) {
    (void)in_sizes; (void)n_in; (void)out_size; (void)ws_size;
    const float* logits = (const float*)d_in[3];   // setup_inputs order
    float* out = (float*)d_out;

    int* partial = (int*)d_ws;                     // [NCHUNKS][64]  (1 MB)
    int* base    = partial + (size_t)NCHUNKS * N_EXPERTS;  // [NCHUNKS][64] (1 MB)

    k1_gate   <<<N_TOKENS / 8, 256, 0, stream>>>(logits, out);
    k2_hist   <<<NCHUNKS,      256, 0, stream>>>(out, partial);
    k3_scan   <<<N_EXPERTS,    256, 0, stream>>>(partial, base, out);
    k4_offsets<<<NCHUNKS,       32, 0, stream>>>(out, base, out);

    const int n4 = (N_TOKENS * N_EXPERTS) / 4;     // 8388608 float4s
    k5_copy   <<<8192, 256, 0, stream>>>((const float4*)logits,
                                         (float4*)(out + OFF_LOGITS), n4);
}
